// SelfAttention_Layer_70102456206152
// MI455X (gfx1250) — matrix operand order repaired and checked by reading, not compiled, not run on hardware
//
#include <hip/hip_runtime.h>
#include <hip/hip_bf16.h>

typedef __attribute__((ext_vector_type(16))) _Float16 v16h;
typedef __attribute__((ext_vector_type(8)))  _Float16 v8h;
typedef __attribute__((ext_vector_type(8)))  float    v8f;
typedef __attribute__((ext_vector_type(4)))  float    v4f;

#define BB 32
#define SS 1024
#define DD 256
#define NEG_INF_F (-4294967295.0f)   // -2^32 + 1

// ---------------- prep kernels ----------------

__global__ void pe_kernel(float* __restrict__ pe) {
  int idx = blockIdx.x * 256 + threadIdx.x;
  if (idx >= SS * DD) return;
  int s = idx >> 8;          // / DD
  int d = idx & (DD - 1);
  float expo = (float)(2 * (d >> 1)) / (float)DD;
  float angle = (float)s * __powf(10000.0f, -expo);
  pe[idx] = (d & 1) ? __cosf(angle) : __sinf(angle);
}

__global__ void wt_kernel(const float* __restrict__ W, _Float16* __restrict__ Wt) {
  int idx = blockIdx.x * 256 + threadIdx.x;
  if (idx >= DD * DD) return;
  int n = idx >> 8;
  int kd = idx & (DD - 1);
  Wt[idx] = (_Float16)W[kd * DD + n];   // Wt[n][k] = W[k][n]
}

// vt[b][d][s] = (f16) v[b][s][d] : makes PV B-fragments contiguous along keys
__global__ void vt_kernel(const float* __restrict__ v, _Float16* __restrict__ vt) {
  int idx = blockIdx.x * 256 + threadIdx.x;           // over B*S*D
  int b = idx >> 18;                                  // / (S*D)
  int rem = idx & (SS * DD - 1);
  int s = rem >> 8;
  int d = rem & (DD - 1);
  vt[((size_t)b * DD + d) * SS + s] = (_Float16)v[idx];
}

__global__ void zero_kernel(float* __restrict__ out) {
  out[blockIdx.x * 256 + threadIdx.x] = 0.0f;
}

// ---------------- projection: xp = relu((x + pe) @ W), f16 output ----------------
// 1 wave per 16x16 output tile. grid.x = (B*S)/16 row tiles, grid.y = D/16 col tiles.

__global__ __launch_bounds__(32) void proj_kernel(const float* __restrict__ x,
                                                  const float* __restrict__ pe,
                                                  const _Float16* __restrict__ Wt,
                                                  _Float16* __restrict__ xp) {
  const int lane = threadIdx.x;
  const int m    = lane & 15;
  const int gr   = lane >> 4;
  const int row0 = blockIdx.x * 16;
  const int col0 = blockIdx.y * 16;
  const int row  = row0 + m;               // A row this lane owns
  const int s    = row & (SS - 1);         // sequence position
  const int n    = col0 + m;               // B column this lane owns

  v8f c = {};
#pragma unroll
  for (int kk = 0; kk < DD; kk += 32) {
    // A fragment: 16-bit A layout -> lane holds K = gr*8+[0..7] and gr*8+16+[0..7]
    const int ka = kk + gr * 8;
    const v4f* xq = (const v4f*)(x  + (size_t)row * DD + ka);
    const v4f* pq = (const v4f*)(pe + (size_t)s   * DD + ka);
    v4f x0 = xq[0], x1 = xq[1], x2 = xq[4], x3 = xq[5];
    v4f p0 = pq[0], p1 = pq[1], p2 = pq[4], p3 = pq[5];
    v16h a;
#pragma unroll
    for (int i = 0; i < 4; ++i) {
      a[i]      = (_Float16)(x0[i] + p0[i]);
      a[i + 4]  = (_Float16)(x1[i] + p1[i]);
      a[i + 8]  = (_Float16)(x2[i] + p2[i]);
      a[i + 12] = (_Float16)(x3[i] + p3[i]);
    }
    // B fragment: lane holds column n, K = kk + gr*8 + [0..7] and kk + gr*8 + 16 + [0..7]
    v16h b;
    ((v8h*)&b)[0] = *(const v8h*)(Wt + (size_t)n * DD + kk + gr * 8);
    ((v8h*)&b)[1] = *(const v8h*)(Wt + (size_t)n * DD + kk + gr * 8 + 16);
    c = __builtin_amdgcn_wmma_f32_16x16x32_f16(false, a, false, b, (short)0, c, false, false);
  }
  // ReLU + store f16 (C layout: VGPR r -> row gr*8+r, col = lane&15)
#pragma unroll
  for (int r = 0; r < 8; ++r) {
    float vv = c[r] > 0.0f ? c[r] : 0.0f;
    xp[(size_t)(row0 + gr * 8 + r) * DD + col0 + m] = (_Float16)vv;
  }
}

// ---------------- fused flash attention + mean reduction ----------------
// 1 wave per (batch, 16-query tile). grid.x = S/16, grid.y = B.

__global__ __launch_bounds__(32) void attn_kernel(const _Float16* __restrict__ qp,
                                                  const _Float16* __restrict__ kp,
                                                  const _Float16* __restrict__ vt,
                                                  const int* __restrict__ mask,
                                                  float* __restrict__ out) {
  __shared__ _Float16 qtile[16 * DD];   // 8 KB
  __shared__ _Float16 ptile[16 * 32];   // 1 KB: probs tile, row-major stride 32

  const int lane = threadIdx.x;
  const int m    = lane & 15;
  const int gr   = lane >> 4;
  const int b    = blockIdx.y;
  const int q0   = blockIdx.x * 16;
  const size_t qbase = ((size_t)b * SS + q0) * DD;

  // stage 16x256 f16 q-tile into LDS; lane copies cols [lane*8, lane*8+8)
#pragma unroll
  for (int r = 0; r < 16; ++r)
    *(v8h*)(qtile + r * DD + lane * 8) = *(const v8h*)(qp + qbase + (size_t)r * DD + lane * 8);
  asm volatile("s_wait_dscnt 0" ::: "memory");

  v8f acc[16] = {};                     // O accumulators: 16 N-tiles x 8 rows
  float mrun[8], lrun[8];
#pragma unroll
  for (int r = 0; r < 8; ++r) { mrun[r] = -3.0e38f; lrun[r] = 0.0f; }

  const float scale = 0.0625f;          // 1/sqrt(256)
  const _Float16* kpb0 = kp + (size_t)b * SS * DD;
  const _Float16* vtb  = vt + (size_t)b * DD * SS;

  for (int key0 = 0; key0 < SS; key0 += 32) {
    // prefetch next chunk's K rows and V columns (global_prefetch_b8)
    if (key0 + 32 < SS) {
      __builtin_prefetch(kpb0 + (size_t)(key0 + 32 + m) * DD + gr * 128, 0, 0);
      __builtin_prefetch(vtb + (size_t)(m * 16 + gr * 8) * SS + key0 + 32, 0, 0);
    }
    // ---- logits: two 16x16 key tiles, K = D = 256 ----
    v8f s0 = {}, s1 = {};
    const _Float16* kpb = kpb0 + (size_t)key0 * DD;
#pragma unroll
    for (int kk = 0; kk < DD; kk += 32) {
      v16h a;
      ((v8h*)&a)[0] = *(const v8h*)(qtile + m * DD + kk + gr * 8);
      ((v8h*)&a)[1] = *(const v8h*)(qtile + m * DD + kk + gr * 8 + 16);
      v16h b0, b1;
      ((v8h*)&b0)[0] = *(const v8h*)(kpb + (size_t)m        * DD + kk + gr * 8);
      ((v8h*)&b0)[1] = *(const v8h*)(kpb + (size_t)m        * DD + kk + gr * 8 + 16);
      ((v8h*)&b1)[0] = *(const v8h*)(kpb + (size_t)(m + 16) * DD + kk + gr * 8);
      ((v8h*)&b1)[1] = *(const v8h*)(kpb + (size_t)(m + 16) * DD + kk + gr * 8 + 16);
      s0 = __builtin_amdgcn_wmma_f32_16x16x32_f16(false, a, false, b0, (short)0, s0, false, false);
      s1 = __builtin_amdgcn_wmma_f32_16x16x32_f16(false, a, false, b1, (short)0, s1, false, false);
    }
    // ---- scale + key mask (C layout: key = lane&15, same for all 8 row-VGPRs) ----
    const int mk0 = mask[b * SS + key0 + m];
    const int mk1 = mask[b * SS + key0 + 16 + m];
#pragma unroll
    for (int r = 0; r < 8; ++r) {
      s0[r] = mk0 ? s0[r] * scale : NEG_INF_F;
      s1[r] = mk1 ? s1[r] * scale : NEG_INF_F;
    }
    // ---- online softmax update (row reductions within 16-lane halves) ----
    float corr[8];
#pragma unroll
    for (int r = 0; r < 8; ++r) {
      float mx = fmaxf(s0[r], s1[r]);
#pragma unroll
      for (int off = 8; off >= 1; off >>= 1)
        mx = fmaxf(mx, __shfl_xor(mx, off, 32));
      float mnew = fmaxf(mrun[r], mx);
      float e0 = __expf(s0[r] - mnew);
      float e1 = __expf(s1[r] - mnew);
      float rs = e0 + e1;
#pragma unroll
      for (int off = 8; off >= 1; off >>= 1)
        rs += __shfl_xor(rs, off, 32);
      float cr = __expf(mrun[r] - mnew);
      lrun[r] = lrun[r] * cr + rs;
      mrun[r] = mnew;
      corr[r] = cr;
      s0[r] = e0;
      s1[r] = e1;
    }
#pragma unroll
    for (int t = 0; t < 16; ++t)
#pragma unroll
      for (int r = 0; r < 8; ++r)
        acc[t][r] *= corr[r];
    // ---- transpose P through LDS into A-fragment layout ----
#pragma unroll
    for (int r = 0; r < 8; ++r) {
      ptile[(gr * 8 + r) * 32 + m]      = (_Float16)s0[r];
      ptile[(gr * 8 + r) * 32 + 16 + m] = (_Float16)s1[r];
    }
    asm volatile("s_wait_dscnt 0" ::: "memory");
    v16h pa;
    ((v8h*)&pa)[0] = *(const v8h*)(ptile + m * 32 + gr * 8);
    ((v8h*)&pa)[1] = *(const v8h*)(ptile + m * 32 + gr * 8 + 16);
    // ---- O += P (16x32) @ V-chunk (32x256): B-frag = two v8h from vt (keys gr*8+[0..7], gr*8+16+[0..7]) ----
#pragma unroll
    for (int t = 0; t < 16; ++t) {
      const int n = t * 16 + m;
      v16h bf;
      ((v8h*)&bf)[0] = *(const v8h*)(vtb + (size_t)n * SS + key0 + gr * 8);
      ((v8h*)&bf)[1] = *(const v8h*)(vtb + (size_t)n * SS + key0 + gr * 8 + 16);
      acc[t] = __builtin_amdgcn_wmma_f32_16x16x32_f16(false, pa, false, bf, (short)0, acc[t], false, false);
    }
  }

  // ---- finalize: out[b,d] += sum_rows(O/l) / S ----
#pragma unroll
  for (int t = 0; t < 16; ++t) {
    float cs = 0.0f;
#pragma unroll
    for (int r = 0; r < 8; ++r) cs += acc[t][r] / lrun[r];
    cs += __shfl_xor(cs, 16, 32);      // add the other 8 rows (lane half swap)
    if (gr == 0)
      atomicAdd(out + b * DD + t * 16 + m, cs * (1.0f / (float)SS));
  }
}

// ---------------- launcher ----------------

extern "C" void kernel_launch(void* const* d_in, const int* in_sizes, int n_in,
                              void* d_out, int out_size, void* d_ws, size_t ws_size,
                              hipStream_t stream) {
  const float* q    = (const float*)d_in[0];
  const float* k    = (const float*)d_in[1];
  const float* v    = (const float*)d_in[2];
  const int*   mask = (const int*)d_in[3];
  const float* W    = (const float*)d_in[4];
  float* out = (float*)d_out;

  char* ws = (char*)d_ws;
  const size_t PE_BYTES = (size_t)SS * DD * sizeof(float);         // 1 MB
  const size_t WT_BYTES = (size_t)DD * DD * sizeof(_Float16);      // 128 KB
  const size_t XP_BYTES = (size_t)BB * SS * DD * sizeof(_Float16); // 16 MB each
  float*    pe = (float*)ws;
  _Float16* Wt = (_Float16*)(ws + PE_BYTES);
  _Float16* qp = (_Float16*)(ws + PE_BYTES + WT_BYTES);
  _Float16* kp = (_Float16*)(ws + PE_BYTES + WT_BYTES + XP_BYTES);
  _Float16* vt = (_Float16*)(ws + PE_BYTES + WT_BYTES + 2 * XP_BYTES);

  pe_kernel<<<(SS * DD) / 256, 256, 0, stream>>>(pe);
  wt_kernel<<<(DD * DD) / 256, 256, 0, stream>>>(W, Wt);
  vt_kernel<<<(BB * SS * DD) / 256, 256, 0, stream>>>(v, vt);
  zero_kernel<<<out_size / 256, 256, 0, stream>>>(out);

  dim3 pgrid((BB * SS) / 16, DD / 16);
  proj_kernel<<<pgrid, 32, 0, stream>>>(q, pe, Wt, qp);
  proj_kernel<<<pgrid, 32, 0, stream>>>(k, pe, Wt, kp);

  dim3 agrid(SS / 16, BB);
  attn_kernel<<<agrid, 32, 0, stream>>>(qp, kp, vt, mask, out);
}
